// MultiHeadSelfAttention_3264175145623
// MI455X (gfx1250) — matrix operand order repaired and checked by reading, not compiled, not run on hardware
//
#include <hip/hip_runtime.h>

// ---------------------------------------------------------------------------
// MHSA with RoPE for MI455X (gfx1250): bf16 WMMA pipeline, register-blocked.
//   d_model=1024, heads=16, d_k=64, b=2, s=2048
// ---------------------------------------------------------------------------

typedef __attribute__((ext_vector_type(16))) __bf16 v16bf;
typedef __attribute__((ext_vector_type(8)))  float  v8f;

union FragBF { v16bf v; uint4 q[2]; };

// A-matrix fragment (16x32 bf16): lane L holds row (L&15).
// lanes 0-15:  K = 0..15 contiguous
// lanes 16-31: K = 16..31 contiguous
__device__ inline v16bf load_a16x32(const __bf16* row, int hi) {
    FragBF f;
    f.q[0] = *(const uint4*)(row + (hi ? 16 : 0));
    f.q[1] = *(const uint4*)(row + 8 + (hi ? 16 : 0));
    return f.v;
}

// B-matrix fragment (32x16 bf16): lane L holds column (L&15).
// lanes 0-15: K = 0..15 contiguous; lanes 16-31: K = 16..31.
__device__ inline v16bf load_b32x16(const __bf16* col, int hi) {
    FragBF f;
    const uint4* p = (const uint4*)(col + (hi ? 16 : 0));
    f.q[0] = p[0];
    f.q[1] = p[1];
    return f.v;
}

__device__ inline v8f wmma_bf16(v16bf a, v16bf b, v8f c) {
    return __builtin_amdgcn_wmma_f32_16x16x32_bf16(
        false, a, false, b, (short)0, c, false, false);
}

__device__ inline float redmax16(float v) {
    v = fmaxf(v, __shfl_xor(v, 1, 16));
    v = fmaxf(v, __shfl_xor(v, 2, 16));
    v = fmaxf(v, __shfl_xor(v, 4, 16));
    v = fmaxf(v, __shfl_xor(v, 8, 16));
    return v;
}
__device__ inline float redsum16(float v) {
    v += __shfl_xor(v, 1, 16);
    v += __shfl_xor(v, 2, 16);
    v += __shfl_xor(v, 4, 16);
    v += __shfl_xor(v, 8, 16);
    return v;
}

// ---------------------------------------------------------------------------
// Kernel A: fp32 -> bf16 conversion
// ---------------------------------------------------------------------------
__global__ void cvt_f32_bf16(const float* __restrict__ in,
                             __bf16* __restrict__ out, int n) {
    int i = blockIdx.x * 256 + threadIdx.x;
    if (i < n) out[i] = (__bf16)in[i];
}

// ---------------------------------------------------------------------------
// Kernel B: QKV projection GEMM (C = X * W^T, f32 acc), 4x4 register-blocked
// (64x64 output per wave), + RoPE epilogue for Q/K, transposed store for V.
//   blockIdx.z: 0=Q, 1=K -> [b][h][s][64] with RoPE ; 2=V -> [b][h][64][s]
// ---------------------------------------------------------------------------
__global__ __launch_bounds__(128) void qkv_rope_kernel(
    const __bf16* __restrict__ Xb,
    const __bf16* __restrict__ Wq, const __bf16* __restrict__ Wk,
    const __bf16* __restrict__ Wv,
    __bf16* __restrict__ Qh, __bf16* __restrict__ Kh,
    __bf16* __restrict__ Vt) {
    const int lane = threadIdx.x & 31;
    const int wave = threadIdx.x >> 5;
    const int hi = lane >> 4;
    const int ln = lane & 15;
    const int n0 = (blockIdx.x * 4 + wave) * 64;   // 64-col block (16 blocks)
    const int m0 = blockIdx.y * 64;                // 64-row block (64 blocks)
    const int which = blockIdx.z;

    const __bf16* W = (which == 0) ? Wq : (which == 1) ? Wk : Wv;

    const __bf16* arow[4];
    const __bf16* bcol[4];
#pragma unroll
    for (int r = 0; r < 4; ++r)
        arow[r] = Xb + (size_t)(m0 + r * 16 + ln) * 1024;
#pragma unroll
    for (int cc = 0; cc < 4; ++cc)
        bcol[cc] = W + (size_t)(n0 + cc * 16 + ln) * 1024;

    v8f acc[4][4];
#pragma unroll
    for (int r = 0; r < 4; ++r)
#pragma unroll
        for (int cc = 0; cc < 4; ++cc) acc[r][cc] = (v8f){};

    for (int k = 0; k < 1024; k += 32) {
        v16bf af[4], bfr[4];
#pragma unroll
        for (int r = 0; r < 4; ++r) af[r] = load_a16x32(arow[r] + k, hi);
#pragma unroll
        for (int cc = 0; cc < 4; ++cc) bfr[cc] = load_b32x16(bcol[cc] + k, hi);
#pragma unroll
        for (int r = 0; r < 4; ++r)
#pragma unroll
            for (int cc = 0; cc < 4; ++cc)
                acc[r][cc] = wmma_bf16(af[r], bfr[cc], acc[r][cc]);
    }

    const int bidx = m0 >> 11;        // batch (2048 rows each)
    const int sbase = m0 & 2047;

    if (which == 2) {
        // V: store transposed per head: Vt[b][h][dd][s]
#pragma unroll
        for (int r = 0; r < 4; ++r)
#pragma unroll
            for (int cc = 0; cc < 4; ++cc)
#pragma unroll
                for (int i = 0; i < 8; ++i) {
                    int s = sbase + r * 16 + i + hi * 8;
                    int e = n0 + cc * 16 + ln;
                    int h = e >> 6, dd = e & 63;
                    Vt[((size_t)(bidx * 16 + h) * 64 + dd) * 2048 + s] =
                        (__bf16)acc[r][cc][i];
                }
    } else {
        __bf16* Dst = (which == 0) ? Qh : Kh;
        const float lg = logf(10000.0f) / 64.0f;
#pragma unroll
        for (int r = 0; r < 4; ++r)
#pragma unroll
            for (int cc = 0; cc < 4; ++cc)
#pragma unroll
                for (int i = 0; i < 8; ++i) {
                    float x = acc[r][cc][i];
                    float xp = __shfl_xor(x, 1, 32);  // RoPE partner column
                    int s = sbase + r * 16 + i + hi * 8;
                    int e = n0 + cc * 16 + ln;
                    int h = e >> 6, dd = e & 63;
                    int j = dd >> 1;
                    float ang = (float)s * __expf(-(float)(2 * j) * lg);
                    float sn, cs;
                    __sincosf(ang, &sn, &cs);
                    float rr = (e & 1) ? fmaf(xp, sn, x * cs)
                                       : fmaf(x, cs, -(xp * sn));
                    Dst[((size_t)(bidx * 16 + h) * 2048 + s) * 64 + dd] =
                        (__bf16)rr;
                }
    }
}

// ---------------------------------------------------------------------------
// Kernel C: causal flash attention, one wave per (b, h, 32-row q block).
// Two 16-row subtiles share the K and V fragments of each 32-wide k-chunk.
//   Q,K: [b][h][2048][64] bf16 ; Vt: [b][h][64][2048] bf16
//   Out Ob: [b][2048][1024] bf16
// ---------------------------------------------------------------------------
__global__ __launch_bounds__(32) void flash_attn_kernel(
    const __bf16* __restrict__ Qh, const __bf16* __restrict__ Kh,
    const __bf16* __restrict__ Vt, __bf16* __restrict__ Ob) {
    __shared__ __align__(16) __bf16 Plds[32 * 32];

    const int lane = threadIdx.x & 31;
    const int hi = lane >> 4;
    const int ln = lane & 15;
    const int q0 = blockIdx.x * 32;
    const int h = blockIdx.y;
    const int b = blockIdx.z;

    const size_t headOff = (size_t)(b * 16 + h) * 2048 * 64;
    const __bf16* Qbase = Qh + headOff;
    const __bf16* Kbase = Kh + headOff;
    const __bf16* Vbase = Vt + headOff;

    v16bf aq[2][2];
#pragma unroll
    for (int t = 0; t < 2; ++t) {
        const __bf16* qrow = Qbase + (size_t)(q0 + t * 16 + ln) * 64;
        aq[t][0] = load_a16x32(qrow, hi);        // d 0..31
        aq[t][1] = load_a16x32(qrow + 32, hi);   // d 32..63
    }

    v8f o[2][4];
    float m[2][8], l[2][8];
#pragma unroll
    for (int t = 0; t < 2; ++t) {
#pragma unroll
        for (int j = 0; j < 4; ++j) o[t][j] = (v8f){};
#pragma unroll
        for (int i = 0; i < 8; ++i) { m[t][i] = -1e30f; l[t][i] = 0.0f; }
    }

    for (int k0 = 0; k0 < q0 + 32; k0 += 32) {
        // ---- shared K fragments for this 32-wide chunk ----
        const __bf16* kc0 = Kbase + (size_t)(k0 + ln) * 64;
        const __bf16* kc1 = Kbase + (size_t)(k0 + 16 + ln) * 64;
        __builtin_prefetch(kc0 + 32 * 64, 0, 0);   // next chunk K rows
        __builtin_prefetch(kc1 + 32 * 64, 0, 0);
        const v16bf bk00 = load_b32x16(kc0, hi);
        const v16bf bk01 = load_b32x16(kc0 + 32, hi);
        const v16bf bk10 = load_b32x16(kc1, hi);
        const v16bf bk11 = load_b32x16(kc1 + 32, hi);

#pragma unroll
        for (int t = 0; t < 2; ++t) {
            v8f s0 = {}, s1 = {};
            s0 = wmma_bf16(aq[t][0], bk00, s0);
            s0 = wmma_bf16(aq[t][1], bk01, s0);
            s1 = wmma_bf16(aq[t][0], bk10, s1);
            s1 = wmma_bf16(aq[t][1], bk11, s1);

            float alpha[8];
#pragma unroll
            for (int i = 0; i < 8; ++i) {
                int qrow_g = q0 + t * 16 + i + hi * 8;
                int c0 = k0 + ln, c1 = k0 + 16 + ln;
                float a0 = s0[i] * 0.125f;   // 1/sqrt(64)
                float a1 = s1[i] * 0.125f;
                if (c0 > qrow_g) a0 = -1e30f;
                if (c1 > qrow_g) a1 = -1e30f;
                float rm = redmax16(fmaxf(a0, a1));
                float mn = fmaxf(m[t][i], rm);
                float al = __expf(m[t][i] - mn);
                float p0 = __expf(a0 - mn);
                float p1 = __expf(a1 - mn);
                l[t][i] = l[t][i] * al + redsum16(p0 + p1);
                m[t][i] = mn;
                alpha[i] = al;
                int row = t * 16 + i + hi * 8;
                Plds[row * 32 + ln] = (__bf16)p0;
                Plds[row * 32 + 16 + ln] = (__bf16)p1;
            }
#pragma unroll
            for (int i = 0; i < 8; ++i) {
                o[t][0][i] *= alpha[i]; o[t][1][i] *= alpha[i];
                o[t][2][i] *= alpha[i]; o[t][3][i] *= alpha[i];
            }
        }
        __syncthreads();   // LDS stores -> A-fragment reload (layout swap)

        const v16bf pa0 = load_a16x32(&Plds[ln * 32], hi);
        const v16bf pa1 = load_a16x32(&Plds[(16 + ln) * 32], hi);
#pragma unroll
        for (int j = 0; j < 4; ++j) {
            const __bf16* vc = Vbase + (size_t)(j * 16 + ln) * 2048 + k0;
            __builtin_prefetch(vc + 32, 0, 0);     // next chunk V run
            const v16bf bv = load_b32x16(vc, hi);
            o[0][j] = wmma_bf16(pa0, bv, o[0][j]);
            o[1][j] = wmma_bf16(pa1, bv, o[1][j]);
        }
        __syncthreads();
    }

    // ---- normalize and store O in [b][s][h*64+dd] layout ----
#pragma unroll
    for (int t = 0; t < 2; ++t)
#pragma unroll
        for (int i = 0; i < 8; ++i) {
            int q = q0 + t * 16 + i + hi * 8;
            float inv = 1.0f / l[t][i];
            size_t base = ((size_t)(b * 2048 + q)) * 1024 + h * 64 + ln;
            Ob[base + 0]  = (__bf16)(o[t][0][i] * inv);
            Ob[base + 16] = (__bf16)(o[t][1][i] * inv);
            Ob[base + 32] = (__bf16)(o[t][2][i] * inv);
            Ob[base + 48] = (__bf16)(o[t][3][i] * inv);
        }
}

// ---------------------------------------------------------------------------
// Kernel D: output projection, out = Ob * Wo^T (f32 result), 4x4 blocked
// ---------------------------------------------------------------------------
__global__ __launch_bounds__(128) void out_gemm_kernel(
    const __bf16* __restrict__ Ob, const __bf16* __restrict__ Wo,
    float* __restrict__ out) {
    const int lane = threadIdx.x & 31;
    const int wave = threadIdx.x >> 5;
    const int hi = lane >> 4;
    const int ln = lane & 15;
    const int n0 = (blockIdx.x * 4 + wave) * 64;
    const int m0 = blockIdx.y * 64;

    const __bf16* arow[4];
    const __bf16* bcol[4];
#pragma unroll
    for (int r = 0; r < 4; ++r)
        arow[r] = Ob + (size_t)(m0 + r * 16 + ln) * 1024;
#pragma unroll
    for (int cc = 0; cc < 4; ++cc)
        bcol[cc] = Wo + (size_t)(n0 + cc * 16 + ln) * 1024;

    v8f acc[4][4];
#pragma unroll
    for (int r = 0; r < 4; ++r)
#pragma unroll
        for (int cc = 0; cc < 4; ++cc) acc[r][cc] = (v8f){};

    for (int k = 0; k < 1024; k += 32) {
        v16bf af[4], bfr[4];
#pragma unroll
        for (int r = 0; r < 4; ++r) af[r] = load_a16x32(arow[r] + k, hi);
#pragma unroll
        for (int cc = 0; cc < 4; ++cc) bfr[cc] = load_b32x16(bcol[cc] + k, hi);
#pragma unroll
        for (int r = 0; r < 4; ++r)
#pragma unroll
            for (int cc = 0; cc < 4; ++cc)
                acc[r][cc] = wmma_bf16(af[r], bfr[cc], acc[r][cc]);
    }

#pragma unroll
    for (int r = 0; r < 4; ++r)
#pragma unroll
        for (int cc = 0; cc < 4; ++cc)
#pragma unroll
            for (int i = 0; i < 8; ++i)
                out[(size_t)(m0 + r * 16 + i + hi * 8) * 1024 +
                    n0 + cc * 16 + ln] = acc[r][cc][i];
}

// ---------------------------------------------------------------------------
// Host launcher
// ---------------------------------------------------------------------------
extern "C" void kernel_launch(void* const* d_in, const int* in_sizes, int n_in,
                              void* d_out, int out_size, void* d_ws, size_t ws_size,
                              hipStream_t stream) {
    const float* x  = (const float*)d_in[0];
    const float* wq = (const float*)d_in[1];
    const float* wk = (const float*)d_in[2];
    const float* wv = (const float*)d_in[3];
    const float* wo = (const float*)d_in[4];
    float* out = (float*)d_out;

    const size_t XN = 2ull * 2048 * 1024;  // 4,194,304
    const size_t WN = 1024ull * 1024;      // 1,048,576

    char* ws = (char*)d_ws;
    __bf16* Xb  = (__bf16*)ws;                     ws += XN * 2;
    __bf16* Wqb = (__bf16*)ws;                     ws += WN * 2;
    __bf16* Wkb = (__bf16*)ws;                     ws += WN * 2;
    __bf16* Wvb = (__bf16*)ws;                     ws += WN * 2;
    __bf16* Wob = (__bf16*)ws;                     ws += WN * 2;
    __bf16* Qh  = (__bf16*)ws;                     ws += XN * 2;
    __bf16* Kh  = (__bf16*)ws;                     ws += XN * 2;
    __bf16* Vt  = (__bf16*)ws;                     ws += XN * 2;
    __bf16* Ob  = (__bf16*)ws;                     ws += XN * 2;

    // 1) fp32 -> bf16
    cvt_f32_bf16<<<(int)((XN + 255) / 256), 256, 0, stream>>>(x, Xb, (int)XN);
    cvt_f32_bf16<<<(int)((WN + 255) / 256), 256, 0, stream>>>(wq, Wqb, (int)WN);
    cvt_f32_bf16<<<(int)((WN + 255) / 256), 256, 0, stream>>>(wk, Wkb, (int)WN);
    cvt_f32_bf16<<<(int)((WN + 255) / 256), 256, 0, stream>>>(wv, Wvb, (int)WN);
    cvt_f32_bf16<<<(int)((WN + 255) / 256), 256, 0, stream>>>(wo, Wob, (int)WN);

    // 2) QKV projections + RoPE (z: 0=Q,1=K,2=V), 64x64 tile per wave
    qkv_rope_kernel<<<dim3(4, 64, 3), 128, 0, stream>>>(
        Xb, Wqb, Wkb, Wvb, Qh, Kh, Vt);

    // 3) causal flash attention (one wave per 32-row q block)
    flash_attn_kernel<<<dim3(64, 16, 2), 32, 0, stream>>>(Qh, Kh, Vt, Ob);

    // 4) output projection, 64x64 tile per wave
    out_gemm_kernel<<<dim3(4, 64), 128, 0, stream>>>(Ob, Wob, out);
}
